// OmegaAttention_16827681865982
// MI455X (gfx1250) — matrix operand order repaired and checked by reading, not compiled, not run on hardware
//
#include <hip/hip_runtime.h>
#include <hip/hip_bf16.h>

// ---------------- problem constants ----------------
constexpr int S  = 4096;
constexpr int H  = 3072;
constexpr int NH = 16;
constexpr int NKV = 8;
constexpr int HD = 256;
constexpr int QKV_D = (NH + 2 * NKV) * HD;   // 8192
constexpr int REP = NH / NKV;                // 2
constexpr float SCALE   = 0.0668153104781061f;   // 224^-0.5
constexpr float SOFTCAP = 50.0f;
constexpr int   WINDOW  = 1024;
constexpr float ROPE_BASE = 1000000.0f;
constexpr float EPS = 1e-6f;

typedef __attribute__((ext_vector_type(16))) _Float16 v16h;
typedef __attribute__((ext_vector_type(8)))  float    v8f;

#define WMMA_F16(a, b, c) \
  __builtin_amdgcn_wmma_f32_16x16x32_f16(false, (a), false, (b), (short)0, (c), false, false)

// ---------------- Tensor Data Mover (gfx1250 TDM) ----------------
#if __has_builtin(__builtin_amdgcn_tensor_load_to_lds)
#define HAVE_TDM 1
#else
#define HAVE_TDM 0
#endif

#if HAVE_TDM
typedef unsigned int u32x4 __attribute__((ext_vector_type(4)));
typedef int          i32x8 __attribute__((ext_vector_type(8)));
typedef int          i32x4 __attribute__((ext_vector_type(4)));

// 2-D tile load via TDM: tile (t1 rows x t0 elems) of f16 from row-major tensor
// (row stride stride0 elems) at gptr -> LDS byte offset lds_off.
// D# packing per CDNA5 ISA section 8 (group0 128b, group1 256b; groups 2/3 zero => 2-D).
__device__ __forceinline__ void tdm_load_2d(const _Float16* gptr, unsigned lds_off,
                                            unsigned d0, unsigned d1, unsigned stride0,
                                            unsigned t0, unsigned t1) {
  unsigned long long ga = (unsigned long long)(uintptr_t)gptr;
  u32x4 g0;
  g0[0] = 1u;                                   // count=1 (valid user descriptor)
  g0[1] = lds_off;                              // lds_addr
  g0[2] = (unsigned)ga;                         // global_addr[31:0]
  g0[3] = (unsigned)(ga >> 32) | (2u << 30);    // global_addr[56:32] | type=2 ("image")
  i32x8 g1;
  g1[0] = 0x00010000;                           // workgroup_mask=0, data_size=1 (2 bytes)
  g1[1] = (int)((d0 & 0xFFFFu) << 16);          // tensor_dim0[15:0]
  g1[2] = (int)((d0 >> 16) | ((d1 & 0xFFFFu) << 16));   // tensor_dim0[31:16] | tensor_dim1[15:0]
  g1[3] = (int)((d1 >> 16) | (t0 << 16));       // tensor_dim1[31:16] | tile_dim0
  g1[4] = (int)(t1 & 0xFFFFu);                  // tile_dim1 | tile_dim2=0
  g1[5] = (int)stride0;                         // tensor_dim0_stride[31:0]
  g1[6] = 0;                                    // stride hi | dim1_stride lo (unused, 2-D)
  g1[7] = 0;
  i32x4 z = {0, 0, 0, 0};
#if defined(__clang_major__) && (__clang_major__ >= 23)
  i32x8 z8 = {0, 0, 0, 0, 0, 0, 0, 0};
  __builtin_amdgcn_tensor_load_to_lds(g0, g1, z, z, z8, 0);
#else
  __builtin_amdgcn_tensor_load_to_lds(g0, g1, z, z, 0);
#endif
}
#endif  // HAVE_TDM

// ---------------- fragment loaders (ISA 7.12.2 layouts, wave32) ----------------
// A 16x32 f16: lane m=lane&15, hi=lane>>4. VGPR i<4: K = hi*8+2i,+1 ; i>=4: K = 16+hi*8+2(i-4),+1.
__device__ __forceinline__ v16h load_a_frag(const _Float16* base, int ld) {
  int lane = threadIdx.x & 31;
  const _Float16* rowp = base + (size_t)(lane & 15) * ld + ((lane >> 4) << 3);
  union { v16h v; uint4 q[2]; } r;
  r.q[0] = *(const uint4*)(rowp);
  r.q[1] = *(const uint4*)(rowp + 16);
  return r.v;
}

// B 32x16 f16: lane n=lane&15, hi=lane>>4. VGPR j<4: K = hi*8+2j,+1 ; j>=4: K = 16+hi*8+2(j-4),+1.  B[kk][n] = base[n*ld + kk].
__device__ __forceinline__ v16h load_b_frag(const _Float16* base, int ld) {
  int lane = threadIdx.x & 31;
  const _Float16* colp = base + (size_t)(lane & 15) * ld + ((lane >> 4) << 3);
  union { v16h v; uint4 q[2]; } r;
  r.q[0] = *(const uint4*)(colp);
  r.q[1] = *(const uint4*)(colp + 16);
  return r.v;
}

// ---------------- f32 -> f16 conversion (8 elems / thread) ----------------
__global__ void cvt_f32_to_f16(const float* __restrict__ src, _Float16* __restrict__ dst, size_t n) {
  size_t i = ((size_t)blockIdx.x * blockDim.x + threadIdx.x) * 8;
  if (i >= n) return;
  float4 a = *(const float4*)(src + i);
  float4 b = *(const float4*)(src + i + 4);
  union { uint4 q; _Float16 e[8]; } r;
  r.e[0] = (_Float16)a.x; r.e[1] = (_Float16)a.y; r.e[2] = (_Float16)a.z; r.e[3] = (_Float16)a.w;
  r.e[4] = (_Float16)b.x; r.e[5] = (_Float16)b.y; r.e[6] = (_Float16)b.z; r.e[7] = (_Float16)b.w;
  *(uint4*)(dst + i) = r.q;
}

// ---------------- GEMM: C[M,N] = A[M,K] * W[N,K]^T ----------------
// 128 threads = 4 waves (2x2), per-wave 32x64, block tile 64x128, K-step 32.
// TDM path: double-buffered LDS tiles, wave0 issues tensor_load_to_lds for tile i+1
// while all waves run WMMA on tile i.
template <bool OUT_HALF>
__global__ __launch_bounds__(128) void gemm_tn(const _Float16* __restrict__ A,
                                               const _Float16* __restrict__ W,
                                               void* __restrict__ Cout,
                                               int M, int N, int K) {
  int t    = threadIdx.x;
  int lane = t & 31;
  int wid  = t >> 5;
  int wm = wid & 1;
  int wn = wid >> 1;
  int m0 = blockIdx.y * 64;
  int n0 = blockIdx.x * 128;

  v8f acc[2][4] = {};

#if HAVE_TDM
  __shared__ alignas(16) _Float16 ldsA[2][64 * 32];
  __shared__ alignas(16) _Float16 ldsW[2][128 * 32];
  const int nt = K / 32;
  if (wid == 0) {
    tdm_load_2d(A + (size_t)m0 * K, (unsigned)(uintptr_t)&ldsA[0][0], K, 64,  K, 32, 64);
    tdm_load_2d(W + (size_t)n0 * K, (unsigned)(uintptr_t)&ldsW[0][0], K, 128, K, 32, 128);
  }
  for (int it = 0; it < nt; ++it) {
    if (wid == 0) __builtin_amdgcn_s_wait_tensorcnt(0);
    __syncthreads();                       // tile it ready; tile it-1 buffer free
    if (wid == 0 && it + 1 < nt) {
      int k1 = (it + 1) * 32;
      tdm_load_2d(A + (size_t)m0 * K + k1, (unsigned)(uintptr_t)&ldsA[(it + 1) & 1][0], K, 64,  K, 32, 64);
      tdm_load_2d(W + (size_t)n0 * K + k1, (unsigned)(uintptr_t)&ldsW[(it + 1) & 1][0], K, 128, K, 32, 128);
    }
    const _Float16* lA = &ldsA[it & 1][0];
    const _Float16* lW = &ldsW[it & 1][0];
    v16h a0 = load_a_frag(lA + (wm * 32)      * 32, 32);
    v16h a1 = load_a_frag(lA + (wm * 32 + 16) * 32, 32);
#pragma unroll
    for (int in = 0; in < 4; ++in) {
      v16h b = load_b_frag(lW + (wn * 64 + 16 * in) * 32, 32);
      acc[0][in] = WMMA_F16(a0, b, acc[0][in]);
      acc[1][in] = WMMA_F16(a1, b, acc[1][in]);
    }
  }
#else
  __shared__ alignas(16) _Float16 ldsA[64 * 32];
  __shared__ alignas(16) _Float16 ldsW[128 * 32];
  for (int k0 = 0; k0 < K; k0 += 32) {
    {
      int row = t >> 1;
      int c   = (t & 1) * 16;
      const uint4* g = (const uint4*)(A + (size_t)(m0 + row) * K + k0 + c);
      uint4* l = (uint4*)(ldsA + row * 32 + c);
      l[0] = g[0];
      l[1] = g[1];
      __builtin_prefetch(A + (size_t)(m0 + row) * K + k0 + 32, 0, 1);
    }
    {
      const uint4* g = (const uint4*)(W + (size_t)(n0 + t) * K + k0);
      uint4* l = (uint4*)(ldsW + t * 32);
      l[0] = g[0]; l[1] = g[1]; l[2] = g[2]; l[3] = g[3];
      __builtin_prefetch(W + (size_t)(n0 + t) * K + k0 + 32, 0, 1);
    }
    __syncthreads();
    v16h a0 = load_a_frag(ldsA + (wm * 32)      * 32, 32);
    v16h a1 = load_a_frag(ldsA + (wm * 32 + 16) * 32, 32);
#pragma unroll
    for (int in = 0; in < 4; ++in) {
      v16h b = load_b_frag(ldsW + (wn * 64 + 16 * in) * 32, 32);
      acc[0][in] = WMMA_F16(a0, b, acc[0][in]);
      acc[1][in] = WMMA_F16(a1, b, acc[1][in]);
    }
    __syncthreads();
  }
#endif

  int col = lane & 15;
  int ro  = (lane >> 4) * 8;     // C layout: lanes16-31 hold rows 8..15
#pragma unroll
  for (int im = 0; im < 2; ++im)
#pragma unroll
    for (int in = 0; in < 4; ++in)
#pragma unroll
      for (int r = 0; r < 8; ++r) {
        size_t row = (size_t)(m0 + wm * 32 + 16 * im + r + ro);
        size_t c   = (size_t)(n0 + wn * 64 + 16 * in + col);
        float val  = acc[im][in][r];
        if (OUT_HALF) ((_Float16*)Cout)[row * N + c] = (_Float16)val;
        else          ((float*)Cout)[row * N + c]    = val;
      }
}

// ---------------- per-(token, head) RMS norm + RoPE ----------------
// grid = (S, NH + 2*NKV), block = 256 (= HD)
__global__ __launch_bounds__(256) void norm_rope_kernel(const _Float16* __restrict__ qkv,
                                                        const float* __restrict__ qnw,
                                                        const float* __restrict__ knw,
                                                        const int* __restrict__ positions,
                                                        _Float16* __restrict__ q,
                                                        _Float16* __restrict__ k,
                                                        _Float16* __restrict__ v) {
  int s  = blockIdx.x;
  int ha = blockIdx.y;
  int t  = threadIdx.x;
  float x = (float)qkv[(size_t)s * QKV_D + (size_t)ha * HD + t];

  if (ha >= NH + NKV) {   // V head: plain copy
    v[((size_t)s * NKV + (ha - NH - NKV)) * HD + t] = (_Float16)x;
    return;
  }

  __shared__ float red[256];
  __shared__ float ybuf[256];
  red[t] = x * x;
  __syncthreads();
#pragma unroll
  for (int off = 128; off > 0; off >>= 1) {
    if (t < off) red[t] += red[t + off];
    __syncthreads();
  }
  float inv = rsqrtf(red[0] * (1.0f / HD) + EPS);
  const float* w = (ha < NH) ? qnw : knw;
  ybuf[t] = x * inv * (1.0f + w[t]);
  __syncthreads();

  float pos = (float)positions[s];
  const int half = HD / 2;
  float out;
  if (t < half) {
    float invf = __powf(ROPE_BASE, -((float)t) / half);
    float fr = pos * invf;
    out = ybuf[t] * __cosf(fr) - ybuf[t + half] * __sinf(fr);
  } else {
    int tt = t - half;
    float invf = __powf(ROPE_BASE, -((float)tt) / half);
    float fr = pos * invf;
    out = ybuf[tt] * __sinf(fr) + ybuf[t] * __cosf(fr);
  }
  if (ha < NH) q[((size_t)s * NH + ha) * HD + t] = (_Float16)(out * SCALE);   // fold score scale into Q
  else         k[((size_t)s * NKV + (ha - NH)) * HD + t] = (_Float16)out;
}

// ---------------- flash attention, sliding window + softcap ----------------
// Block = 4 waves = 64 consecutive Q rows of one head; shared 32-key K/V tiles in LDS.
// K tile staged by TDM (wave0 issues, s_wait_tensorcnt), V tile transposed manually.
constexpr int AW = 4;
__global__ __launch_bounds__(32 * AW) void attn_kernel(const _Float16* __restrict__ Q,
                                                       const _Float16* __restrict__ Kc,
                                                       const _Float16* __restrict__ Vc,
                                                       _Float16* __restrict__ Oc) {
  __shared__ alignas(16) _Float16 ldsK[32 * 256];     // [key][d]
  __shared__ alignas(16) _Float16 ldsVT[256 * 32];    // [d][key] (transposed)
  __shared__ alignas(16) _Float16 lds_p[AW][16 * 32]; // per-wave P tile

  int t    = threadIdx.x;
  int lane = t & 31;
  int wid  = t >> 5;
  int h    = blockIdx.y;
  int qsb  = blockIdx.x * 64;
  int qs   = qsb + wid * 16;
  int kvh  = h / REP;

  // Q fragments (16 rows x 256 dims, 8 K-chunks of 32); SCALE folded in upstream
  v16h qa[8];
  const _Float16* qbase = Q + ((size_t)qs * NH + h) * HD;
#pragma unroll
  for (int c = 0; c < 8; ++c) qa[c] = load_a_frag(qbase + c * 32, NH * HD);

  v8f acc[16] = {};                       // O: 16 rows x 256 dims fp32
  float m_i[8], l_i[8];
#pragma unroll
  for (int r = 0; r < 8; ++r) { m_i[r] = -3.0e38f; l_i[r] = 0.0f; }

  int col = lane & 15;
  int ro  = (lane >> 4) * 8;

  int lo = qsb - (WINDOW - 1);
  if (lo < 0) lo = 0;
  lo &= ~31;

  for (int kt2 = lo; kt2 <= qsb + 63; kt2 += 32) {
    // ---- stage K tile (TDM if available), V tile transposed by all threads ----
#if HAVE_TDM
    if (wid == 0) {
      tdm_load_2d(Kc + ((size_t)kt2 * NKV + kvh) * HD, (unsigned)(uintptr_t)&ldsK[0],
                  HD, 32, NKV * HD, HD, 32);
    }
#else
    {
      int key = t >> 2;
      int ch  = (t & 3) * 64;
      const uint4* g = (const uint4*)(Kc + ((size_t)(kt2 + key) * NKV + kvh) * HD + ch);
      uint4* l = (uint4*)(ldsK + key * 256 + ch);
#pragma unroll
      for (int i = 0; i < 8; ++i) l[i] = g[i];
    }
#endif
    {
      int key = t >> 2;
      int ch  = (t & 3) * 64;
      const _Float16* gv = Vc + ((size_t)(kt2 + key) * NKV + kvh) * HD + ch;
      union { uint4 q; _Float16 e[8]; } buf;
#pragma unroll
      for (int i = 0; i < 8; ++i) {
        buf.q = ((const uint4*)gv)[i];
#pragma unroll
        for (int j = 0; j < 8; ++j) ldsVT[(size_t)(ch + i * 8 + j) * 32 + key] = buf.e[j];
      }
    }
#if HAVE_TDM
    if (wid == 0) __builtin_amdgcn_s_wait_tensorcnt(0);
#endif
    __syncthreads();

    // ---- scores for 32 keys: two 16x16 C tiles ----
    v8f s0 = {}, s1 = {};
#pragma unroll
    for (int c = 0; c < 8; ++c) {
      v16h b0 = load_b_frag(ldsK + c * 32,            256);
      v16h b1 = load_b_frag(ldsK + 16 * 256 + c * 32, 256);
      s0 = WMMA_F16(qa[c], b0, s0);
      s1 = WMMA_F16(qa[c], b1, s1);
    }

    // ---- softcap + window mask + online softmax ----
    float p0[8], p1[8], alpha[8];
#pragma unroll
    for (int r = 0; r < 8; ++r) {
      int row = qs + r + ro;
      float v0 = SOFTCAP * tanhf(s0[r] * (1.0f / SOFTCAP));
      float v1 = SOFTCAP * tanhf(s1[r] * (1.0f / SOFTCAP));
      int j0 = kt2 + col;
      int j1 = kt2 + 16 + col;
      if (!((j0 <= row) && (row - j0 < WINDOW))) v0 = -3.0e38f;
      if (!((j1 <= row) && (row - j1 < WINDOW))) v1 = -3.0e38f;

      float mt = fmaxf(v0, v1);
#pragma unroll
      for (int off = 1; off < 16; off <<= 1) mt = fmaxf(mt, __shfl_xor(mt, off, 32));
      float m_new = fmaxf(m_i[r], mt);
      float a  = (m_i[r] > -1.0e37f) ? __expf(m_i[r] - m_new) : 0.0f;
      float e0 = (v0   > -1.0e37f) ? __expf(v0 - m_new) : 0.0f;
      float e1 = (v1   > -1.0e37f) ? __expf(v1 - m_new) : 0.0f;
      float ts = e0 + e1;
#pragma unroll
      for (int off = 1; off < 16; off <<= 1) ts += __shfl_xor(ts, off, 32);
      l_i[r] = l_i[r] * a + ts;
      m_i[r] = m_new;
      alpha[r] = a;
      p0[r] = e0;
      p1[r] = e1;
    }

    // rescale accumulated output
#pragma unroll
    for (int cc = 0; cc < 16; ++cc)
#pragma unroll
      for (int r = 0; r < 8; ++r) acc[cc][r] *= alpha[r];

    // ---- stage P (C layout) to LDS, reread as A fragment (wave-local) ----
    _Float16* pl = &lds_p[wid][0];
#pragma unroll
    for (int r = 0; r < 8; ++r) {
      pl[(r + ro) * 32 + col]      = (_Float16)p0[r];
      pl[(r + ro) * 32 + 16 + col] = (_Float16)p1[r];
    }
    asm volatile("s_wait_dscnt 0" ::: "memory");
    v16h pa = load_a_frag(pl, 32);

    // ---- P @ V : 16 d-chunks from transposed V tile (contiguous b128) ----
#pragma unroll
    for (int cc = 0; cc < 16; ++cc) {
      v16h bv = load_b_frag(ldsVT + (size_t)(cc * 16) * 32, 32);
      acc[cc] = WMMA_F16(pa, bv, acc[cc]);
    }
    __syncthreads();   // protect ldsK/ldsVT before next tile overwrites
  }

  // ---- epilogue: normalize and store attn output [S, NH*HD] f16 ----
#pragma unroll
  for (int r = 0; r < 8; ++r) {
    float invl = 1.0f / l_i[r];
    size_t row = (size_t)(qs + r + ro);
#pragma unroll
    for (int cc = 0; cc < 16; ++cc) {
      Oc[row * (NH * HD) + (size_t)h * HD + cc * 16 + col] = (_Float16)(acc[cc][r] * invl);
    }
  }
}

// ---------------- host-side orchestration ----------------
extern "C" void kernel_launch(void* const* d_in, const int* in_sizes, int n_in,
                              void* d_out, int out_size, void* d_ws, size_t ws_size,
                              hipStream_t stream) {
  const float* hs   = (const float*)d_in[0];
  const int*   pos  = (const int*)d_in[1];
  const float* qkvw = (const float*)d_in[2];
  const float* ow   = (const float*)d_in[3];
  const float* qnw  = (const float*)d_in[4];
  const float* knw  = (const float*)d_in[5];
  float* out = (float*)d_out;

  _Float16* p = (_Float16*)d_ws;
  _Float16* hsh   = p; p += (size_t)S * H;
  _Float16* wqkvh = p; p += (size_t)QKV_D * H;
  _Float16* woh   = p; p += (size_t)H * (NH * HD);
  _Float16* qkvh  = p; p += (size_t)S * QKV_D;
  _Float16* qh    = p; p += (size_t)S * NH * HD;
  _Float16* kh    = p; p += (size_t)S * NKV * HD;
  _Float16* vh    = p; p += (size_t)S * NKV * HD;
  _Float16* attnh = p; p += (size_t)S * NH * HD;

  // 1) downconvert activations + weights to f16
  {
    size_t n = (size_t)S * H;
    cvt_f32_to_f16<<<(unsigned)((n / 8 + 255) / 256), 256, 0, stream>>>(hs, hsh, n);
    n = (size_t)QKV_D * H;
    cvt_f32_to_f16<<<(unsigned)((n / 8 + 255) / 256), 256, 0, stream>>>(qkvw, wqkvh, n);
    n = (size_t)H * (NH * HD);
    cvt_f32_to_f16<<<(unsigned)((n / 8 + 255) / 256), 256, 0, stream>>>(ow, woh, n);
  }

  // 2) QKV projection: [S,H] x [QKV_D,H]^T -> [S,QKV_D] (f16)
  {
    dim3 grid(QKV_D / 128, S / 64);
    gemm_tn<true><<<grid, 128, 0, stream>>>(hsh, wqkvh, (void*)qkvh, S, QKV_D, H);
  }

  // 3) RMS norm + RoPE, split into q/k/v (score scale folded into q)
  {
    dim3 grid(S, NH + 2 * NKV);
    norm_rope_kernel<<<grid, 256, 0, stream>>>(qkvh, qnw, knw, pos, qh, kh, vh);
  }

  // 4) sliding-window flash attention with softcap
  {
    dim3 grid(S / 64, NH);
    attn_kernel<<<grid, 32 * AW, 0, stream>>>(qh, kh, vh, attnh);
  }

  // 5) output projection: [S, NH*HD] x [H, NH*HD]^T -> [S,H] (f32 out)
  {
    dim3 grid(H / 128, S / 64);
    gemm_tn<false><<<grid, 128, 0, stream>>>(attnh, woh, (void*)out, S, H, NH * HD);
  }
}
